// MaskedContrastiveLoss_30571577213285
// MI455X (gfx1250) — compile-verified, matrix operand order checked by reading, not run on hardware
//
#include <hip/hip_runtime.h>
#include <math.h>

// MI455X / gfx1250: wave32, WMMA bf16 16x16x32, async global->LDS staging.
typedef __attribute__((ext_vector_type(16))) __bf16 v16bf;
typedef __attribute__((ext_vector_type(8)))  float  v8f;
typedef __attribute__((ext_vector_type(4)))  float  f32x4;
typedef int i32x4 __attribute__((vector_size(16)));

#define BSZ   256
#define DSZ   65536
#define CHUNK 256                       // K elements per workgroup (split-K)
#define NSTEP (CHUNK / 32)              // 8 WMMA K-steps per workgroup
#define ROWB  144                       // padded LDS row stride (32 floats + 16B pad)
#define TILEB (256 * ROWB)              // one staged 256x32 f32 tile = 36 KB
#define SMEMB (4 * TILEB)               // A/B double-buffered = 144 KB
#define TEMPERATURE 14.285714285714286f // 1/0.07

// ---- async global->LDS copy (16 B per lane), ASYNCcnt-tracked -------------
// Probe-discovered signature: pointer params are 'int __vector_size__(16) *'.
#if __has_builtin(__builtin_amdgcn_global_load_async_to_lds_b128)
#define ASYNC_CP16(lp, gp)                                                    \
    __builtin_amdgcn_global_load_async_to_lds_b128(                           \
        (i32x4*)(gp), (i32x4*)(lp), 0, 0)
#else
#define ASYNC_CP16(lp, gp)                                                    \
    asm volatile("global_load_async_to_lds_b128 %0, %1, off"                  \
                 :: "v"((unsigned)(unsigned long long)(lp)),                  \
                    "v"((unsigned long long)(gp))                             \
                 : "memory")
#endif

#if __has_builtin(__builtin_amdgcn_s_wait_asynccnt)
#define WAIT_ASYNC() __builtin_amdgcn_s_wait_asynccnt(0)
#else
#define WAIT_ASYNC() asm volatile("s_wait_asynccnt 0x0" ::: "memory")
#endif

// ---------------------------------------------------------------------------
// Read one 16x32 bf16 WMMA operand fragment from a staged LDS tile.
// ISA layout (16-bit A/B operand), lane -> one row of the operand:
//   lanes 0-15 : elems 0..7 -> K +0..7,   elems 8..15 -> K +16..23
//   lanes 16-31: elems 0..7 -> K +8..15,  elems 8..15 -> K +24..31
// ---------------------------------------------------------------------------
__device__ __forceinline__ v16bf lds_frag_bf16(const unsigned char* bufbase,
                                               int row, int half) {
    const float* p0 = (const float*)(bufbase + row * ROWB) + half * 8;
    f32x4 a0 = *(const f32x4*)(p0);
    f32x4 a1 = *(const f32x4*)(p0 + 4);
    f32x4 a2 = *(const f32x4*)(p0 + 16);
    f32x4 a3 = *(const f32x4*)(p0 + 20);
    v16bf r;
    r[0]  = (__bf16)a0[0]; r[1]  = (__bf16)a0[1]; r[2]  = (__bf16)a0[2]; r[3]  = (__bf16)a0[3];
    r[4]  = (__bf16)a1[0]; r[5]  = (__bf16)a1[1]; r[6]  = (__bf16)a1[2]; r[7]  = (__bf16)a1[3];
    r[8]  = (__bf16)a2[0]; r[9]  = (__bf16)a2[1]; r[10] = (__bf16)a2[2]; r[11] = (__bf16)a2[3];
    r[12] = (__bf16)a3[0]; r[13] = (__bf16)a3[1]; r[14] = (__bf16)a3[2]; r[15] = (__bf16)a3[3];
    return r;
}

// Stage one 256-row x 32-float slice (row-padded) into LDS: 4 async b128/thread.
__device__ __forceinline__ void stage_tile(const float* __restrict__ gsrc,
                                           unsigned char* ldsbase, int tid) {
#pragma unroll
    for (int i = 0; i < 4; ++i) {
        int c   = tid + 512 * i;   // 2048 chunks of 16 B
        int row = c >> 3;          // 8 chunks per 128-B row
        int col = c & 7;
        const float* gp = gsrc + (long long)row * DSZ + col * 4;
        unsigned char* lp = ldsbase + row * ROWB + col * 16;
        ASYNC_CP16(lp, gp);
    }
}

// ---------------------------------------------------------------------------
// Kernel 0: zero the 256x256 f32 accumulator in workspace.
// ---------------------------------------------------------------------------
__global__ __launch_bounds__(256) void zero_logits(float* __restrict__ logits) {
    int idx = blockIdx.x * 256 + threadIdx.x;
    f32x4 z = {0.f, 0.f, 0.f, 0.f};
    *(f32x4*)(logits + idx * 4) = z;
}

// ---------------------------------------------------------------------------
// Kernel 1: split-K GEMM with async-LDS double buffering.
// 512 threads = 16 waves (4x4 grid); wave block = 64x64 = 4x4 WMMA tiles.
// LDS layout: [buf (2)][A tile | B tile], each tile 36 KB, total 144 KB.
// ---------------------------------------------------------------------------
__global__ __launch_bounds__(512) void gemm_splitk(const float* __restrict__ A,
                                                   const float* __restrict__ Bm,
                                                   float* __restrict__ logits) {
    extern __shared__ unsigned char smem[];

    const int tid    = threadIdx.x;
    const int wave   = tid >> 5;
    const int lane   = tid & 31;
    const int lane15 = lane & 15;
    const int half   = lane >> 4;
    const int rowbase = (wave >> 2) * 64;
    const int colbase = (wave & 3) * 64;
    const long long k0 = (long long)blockIdx.x * CHUNK;

    const float* Abase = A + k0;
    const float* Bbase = Bm + k0;

    // prologue: stage buffer 0 (runtime pointer arithmetic only; no pointer
    // arrays -- aggregate initializers of LDS-derived pointers break ld.lld)
    stage_tile(Abase, smem, tid);
    stage_tile(Bbase, smem + TILEB, tid);

    v8f acc[4][4] = {};

#pragma unroll 1
    for (int s = 0; s < NSTEP; ++s) {
        WAIT_ASYNC();          // my async stores for buf[s&1] have landed
        __syncthreads();       // everyone staged buf[s&1]; everyone done reading buf[(s+1)&1]
        if (s + 1 < NSTEP) {
            unsigned char* nxt = smem + ((s + 1) & 1) * (2 * TILEB);
            stage_tile(Abase + (s + 1) * 32, nxt, tid);
            stage_tile(Bbase + (s + 1) * 32, nxt + TILEB, tid);
        }
        const unsigned char* As = smem + (s & 1) * (2 * TILEB);
        const unsigned char* Bs = As + TILEB;

        v16bf af[4];
#pragma unroll
        for (int m = 0; m < 4; ++m)
            af[m] = lds_frag_bf16(As, rowbase + 16 * m + lane15, half);
#pragma unroll
        for (int n = 0; n < 4; ++n) {
            v16bf b = lds_frag_bf16(Bs, colbase + 16 * n + lane15, half);
#pragma unroll
            for (int m = 0; m < 4; ++m)
                acc[m][n] = __builtin_amdgcn_wmma_f32_16x16x32_bf16(
                    /*neg_a=*/false, af[m], /*neg_b=*/false, b,
                    /*c_mod=*/(short)0, acc[m][n],
                    /*reuse_a=*/false, /*reuse_b=*/false);
        }
    }

    // C/D layout: lanes 0-15 -> N=lane, M = vgpr; lanes 16-31 -> N=lane-16, M = 8+vgpr.
#pragma unroll
    for (int m = 0; m < 4; ++m)
#pragma unroll
        for (int n = 0; n < 4; ++n)
#pragma unroll
            for (int v = 0; v < 8; ++v) {
                int R = rowbase + 16 * m + half * 8 + v;
                int C = colbase + 16 * n + lane15;
                atomicAdd(&logits[R * BSZ + C], acc[m][n][v]);
            }
}

// ---------------------------------------------------------------------------
// Kernel 2: row/col logsumexp + diagonal -> scalar loss. One 256-thread block.
// ---------------------------------------------------------------------------
__global__ __launch_bounds__(256) void lse_loss(const float* __restrict__ logits,
                                                float* __restrict__ out) {
    const int t = threadIdx.x;

    float mr = -INFINITY;
    for (int j = 0; j < BSZ; ++j) mr = fmaxf(mr, logits[t * BSZ + j] * TEMPERATURE);
    float sr = 0.f;
    for (int j = 0; j < BSZ; ++j) sr += expf(logits[t * BSZ + j] * TEMPERATURE - mr);
    float lse_row = mr + logf(sr);

    float mc = -INFINITY;
    for (int i = 0; i < BSZ; ++i) mc = fmaxf(mc, logits[i * BSZ + t] * TEMPERATURE);
    float sc = 0.f;
    for (int i = 0; i < BSZ; ++i) sc += expf(logits[i * BSZ + t] * TEMPERATURE - mc);
    float lse_col = mc + logf(sc);

    float pos = logits[t * BSZ + t] * TEMPERATURE;
    float contrib = (lse_row - pos) + (lse_col - pos);

    __shared__ float red[256];
    red[t] = contrib;
    __syncthreads();
    for (int off = 128; off > 0; off >>= 1) {
        if (t < off) red[t] += red[t + off];
        __syncthreads();
    }
    if (t == 0) out[0] = red[0] * 0.5f / (float)BSZ;
}

// ---------------------------------------------------------------------------
extern "C" void kernel_launch(void* const* d_in, const int* in_sizes, int n_in,
                              void* d_out, int out_size, void* d_ws, size_t ws_size,
                              hipStream_t stream) {
    const float* feat_A = (const float*)d_in[0];
    const float* feat_B = (const float*)d_in[1];
    float* logits = (float*)d_ws;            // 256*256*4 = 256 KB of workspace
    float* out    = (float*)d_out;

    // 144 KB dynamic LDS (> 64 KB static limit); deterministic, capture-safe.
    (void)hipFuncSetAttribute((const void*)gemm_splitk,
                              hipFuncAttributeMaxDynamicSharedMemorySize, SMEMB);

    zero_logits<<<64, 256, 0, stream>>>(logits);
    gemm_splitk<<<DSZ / CHUNK, 512, SMEMB, stream>>>(feat_A, feat_B, logits);
    lse_loss<<<1, 256, 0, stream>>>(logits, out);
}
